// AttentionLayer_35527969472786
// MI455X (gfx1250) — matrix operand order repaired and checked by reading, not compiled, not run on hardware
//
#include <hip/hip_runtime.h>
#include <math.h>

// ---------------------------------------------------------------------------
// Bahdanau attention, MI455X (gfx1250, wave32, WMMA).
// B=64, S=4096, ENC=512, A=256.
//   K0: dec_proj[b,a]   = decoder_hidden @ W_dec                (tiny)
//   K1: scores[b,s]     = v . tanh(enc @ W_enc + dec_proj)      (WMMA bf16, fused)
//   K2: weights         = softmax(scores) over S, in place      (bandwidth trivial)
//   K3: context[b,e]    = sum_s weights * encoder_out           (bandwidth bound)
// Roofline: 2x 512MiB encoder reads / 23.3 TB/s ~= 45us floor; bf16 WMMA GEMM
// (~4.2M v_wmma, 8192 MAC each) lands at a similar ~25-30us, balanced.
// ---------------------------------------------------------------------------

#define BATCH  64
#define S_LEN  4096
#define ENC_D  512
#define A_D    256
#define KT     64            // K columns of W staged in LDS per phase
#define LDSK   (KT + 8)      // padded row stride (ushorts) -> spread LDS banks

typedef __attribute__((ext_vector_type(16))) __bf16 v16bf;
typedef __attribute__((ext_vector_type(8)))  float  v8f;
typedef __attribute__((ext_vector_type(4)))  float  v4f;

__device__ __forceinline__ unsigned short f2bf(float f) {
    union { float f; unsigned int u; } c; c.f = f;
    unsigned int r = c.u + 0x7FFFu + ((c.u >> 16) & 1u);   // round-to-nearest-even
    return (unsigned short)(r >> 16);
}

// Hardware transcendental tanh (V_TANH_F32, gfx1250 TRANS op) instead of the
// ~25-op OCML expansion: the epilogue does 67M tanh and would otherwise rival
// the WMMA loop itself in cycles.
__device__ __forceinline__ float fast_tanh(float x) {
#if __has_builtin(__builtin_amdgcn_tanhf)
    return __builtin_amdgcn_tanhf(x);
#else
    float y;
    asm volatile("v_tanh_f32 %0, %1" : "=v"(y) : "v"(x));
    return y;
#endif
}

// ---------------------------------------------------------------------------
// K0: dec_proj = decoder_hidden[B,512] @ W_dec[512,256] -> ws[B,256]
// ---------------------------------------------------------------------------
__global__ __launch_bounds__(256)
void k_decproj(const float* __restrict__ dh, const float* __restrict__ Wdec,
               float* __restrict__ decp)
{
    const int b = blockIdx.x;
    const int a = threadIdx.x;
    float acc = 0.f;
    for (int k = 0; k < ENC_D; k += 4) {
#pragma unroll
        for (int j = 0; j < 4; ++j)
            acc = fmaf(dh[b * ENC_D + k + j], Wdec[(size_t)(k + j) * A_D + a], acc);
    }
    decp[b * A_D + a] = acc;
}

// ---------------------------------------------------------------------------
// K1: fused energy-GEMM + tanh + dot(v) -> scores (written into weights slot)
// Workgroup: 256 threads = 8 waves; each wave: 16 rows x all 256 cols.
// W_enc staged transposed as bf16 in LDS (KT-column phases).
// ---------------------------------------------------------------------------
__global__ __launch_bounds__(256)
void k_scores(const float* __restrict__ enc,  const float* __restrict__ Wenc,
              const float* __restrict__ decp, const float* __restrict__ v,
              float* __restrict__ scores)
{
    __shared__ __attribute__((aligned(32))) unsigned short ldsW[A_D * LDSK]; // 36 KB

    const int tid   = threadIdx.x;
    const int wave  = tid >> 5;
    const int lane  = tid & 31;
    const int nl    = lane & 15;     // row (A-frag) / N column (B-frag) within tile
    const int khalf = lane >> 4;     // K-half selector (this code's order: k = 16*khalf + s)
    const int wgRow0 = blockIdx.x * 128;          // 128 rows per WG, 1 batch per WG
    const int row0   = wgRow0 + wave * 16;
    const int b      = wgRow0 / S_LEN;

    v8f acc[16] = {};                              // 16x16 f32 C tiles covering N=256

    const float* arow = enc + (size_t)(row0 + nl) * ENC_D;

    for (int k0 = 0; k0 < ENC_D; k0 += KT) {
        __syncthreads();
        // stage W_enc[k0:k0+KT, 0:256] -> ldsW[n][k] as bf16 (transpose + convert)
        for (int idx = tid; idx < KT * A_D; idx += 256) {
            const int kk = idx >> 8;               // 0..KT-1
            const int nn = idx & 255;              // 0..255
            ldsW[nn * LDSK + kk] = f2bf(Wenc[(size_t)(k0 + kk) * A_D + nn]);
        }
        __syncthreads();

#pragma unroll
        for (int kk = 0; kk < KT; kk += 32) {
            // A fragment (16x32 bf16): lane half 0 -> K 0..15, half 1 -> K 16..31
            const float* ap = arow + k0 + kk + khalf * 16;
            const v4f f0 = *(const v4f*)(ap);
            const v4f f1 = *(const v4f*)(ap + 4);
            const v4f f2 = *(const v4f*)(ap + 8);
            const v4f f3 = *(const v4f*)(ap + 12);
            v16bf afrag;
#pragma unroll
            for (int i = 0; i < 4; ++i) {
                afrag[i]      = (__bf16)f0[i];
                afrag[4 + i]  = (__bf16)f1[i];
                afrag[8 + i]  = (__bf16)f2[i];
                afrag[12 + i] = (__bf16)f3[i];
            }
#pragma unroll
            for (int nt = 0; nt < 16; ++nt) {
                // B fragment (32x16 bf16): lane = N, lanes 0-15 K 0..15, 16-31 K 16..31
                const v16bf bfrag = *(const v16bf*)
                    (&ldsW[(nt * 16 + nl) * LDSK + kk + khalf * 16]);
                acc[nt] = __builtin_amdgcn_wmma_f32_16x16x32_bf16(
                    false, afrag, false, bfrag, (short)0, acc[nt], false, false);
            }
        }
    }

    // Epilogue: scores[m] = sum_a v[a] * tanh(acc[m,a] + dec_proj[b,a])
    // C layout: vgpr r, lanes 0-15 -> (M=r, N=lane); lanes 16-31 -> (M=r+8, N=lane-16)
    float p[8];
#pragma unroll
    for (int r = 0; r < 8; ++r) p[r] = 0.f;
#pragma unroll
    for (int nt = 0; nt < 16; ++nt) {
        const int acol = nt * 16 + nl;
        const float dp = decp[b * A_D + acol];
        const float vv = v[acol];
#pragma unroll
        for (int r = 0; r < 8; ++r)
            p[r] += vv * fast_tanh(acc[nt][r] + dp);
    }
    // reduce across the 16 lanes of each half (xor stays within the group)
#pragma unroll
    for (int r = 0; r < 8; ++r) {
        p[r] += __shfl_xor(p[r], 1);
        p[r] += __shfl_xor(p[r], 2);
        p[r] += __shfl_xor(p[r], 4);
        p[r] += __shfl_xor(p[r], 8);
    }
    if (nl < 8)
        scores[row0 + khalf * 8 + nl] = p[nl];
}

// ---------------------------------------------------------------------------
// K2: per-batch softmax over S=4096, in place in d_out weights region
// ---------------------------------------------------------------------------
__global__ __launch_bounds__(256)
void k_softmax(float* __restrict__ wts, const unsigned char* __restrict__ mask)
{
    __shared__ float red[8];
    const int b   = blockIdx.x;
    const int tid = threadIdx.x;
    float* w = wts + (size_t)b * S_LEN;
    const unsigned char* m = mask + (size_t)b * S_LEN;

    float vals[16];
    float mx = -__builtin_inff();
#pragma unroll
    for (int i = 0; i < 16; ++i) {
        const int s = tid + i * 256;
        float x = w[s];
        if (m[s]) x = -__builtin_inff();
        vals[i] = x;
        mx = fmaxf(mx, x);
    }
    for (int o = 1; o < 32; o <<= 1) mx = fmaxf(mx, __shfl_xor(mx, o));
    if ((tid & 31) == 0) red[tid >> 5] = mx;
    __syncthreads();
    float bmx = red[0];
#pragma unroll
    for (int i = 1; i < 8; ++i) bmx = fmaxf(bmx, red[i]);

    float sum = 0.f;
#pragma unroll
    for (int i = 0; i < 16; ++i) { vals[i] = __expf(vals[i] - bmx); sum += vals[i]; }
    for (int o = 1; o < 32; o <<= 1) sum += __shfl_xor(sum, o);
    __syncthreads();
    if ((tid & 31) == 0) red[tid >> 5] = sum;
    __syncthreads();
    float bs = 0.f;
#pragma unroll
    for (int i = 0; i < 8; ++i) bs += red[i];
    const float inv = 1.0f / bs;
#pragma unroll
    for (int i = 0; i < 16; ++i) w[tid + i * 256] = vals[i] * inv;
}

// ---------------------------------------------------------------------------
// K3: context[b,e] = sum_s weights[b,s] * enc[b,s,e]   (HBM-bound GEMV)
// grid (ENC/64, B), block 64
// ---------------------------------------------------------------------------
__global__ __launch_bounds__(64)
void k_context(const float* __restrict__ enc, const float* __restrict__ wts,
               float* __restrict__ ctx)
{
    const int b = blockIdx.y;
    const int e = blockIdx.x * 64 + threadIdx.x;
    const float* ep = enc + (size_t)b * S_LEN * ENC_D + e;
    const float* wp = wts + (size_t)b * S_LEN;
    float acc = 0.f;
    for (int s = 0; s < S_LEN; s += 8) {
#pragma unroll
        for (int j = 0; j < 8; ++j)
            acc = fmaf(wp[s + j], ep[(size_t)(s + j) * ENC_D], acc);
    }
    ctx[b * ENC_D + e] = acc;
}

// ---------------------------------------------------------------------------
extern "C" void kernel_launch(void* const* d_in, const int* in_sizes, int n_in,
                              void* d_out, int out_size, void* d_ws, size_t ws_size,
                              hipStream_t stream)
{
    const float*         enc  = (const float*)d_in[0];          // [64,4096,512]
    const float*         dh   = (const float*)d_in[1];          // [64,512]
    const unsigned char* mask = (const unsigned char*)d_in[2];  // [64,4096] bool
    const float*         Wenc = (const float*)d_in[3];          // [512,256]
    const float*         Wdec = (const float*)d_in[4];          // [512,256]
    const float*         v    = (const float*)d_in[5];          // [256]

    float* out  = (float*)d_out;
    float* ctx  = out;                         // [64,512]   first output
    float* wts  = out + BATCH * ENC_D;         // [64,4096]  second output
    float* decp = (float*)d_ws;                // [64,256] scratch (64 KB)

    k_decproj<<<BATCH, 256, 0, stream>>>(dh, Wdec, decp);
    k_scores <<<(BATCH * S_LEN) / 128, 256, 0, stream>>>(enc, Wenc, decp, v, wts);
    k_softmax<<<BATCH, 256, 0, stream>>>(wts, mask);
    dim3 g3(ENC_D / 64, BATCH);
    k_context<<<g3, 64, 0, stream>>>(enc, wts, ctx);
}
